// MultiHeadAttention_90881507984168
// MI455X (gfx1250) — matrix operand order repaired and checked by reading, not compiled, not run on hardware
//
#include <hip/hip_runtime.h>
#include <math.h>

// ---------------------------------------------------------------------------
// MI455X (gfx1250) fused multi-head attention, wave32 + WMMA f16 (f32 accum),
// TDM (tensor_load_to_lds) double-buffered K/V staging, NT cache policy on the
// single-use bias/attn streams.
// Pipeline:
//   1) proj_gemm<float,0>   : qh/kh/vh = X @ W + b   (f32 in -> f16 [B,H,L,D])
//   2) attn_fused           : S = qh kh^T/32 + bias, mask, exact softmax
//                             (full 32x2048 f32 score rows resident in LDS),
//                             writes attn (f32, NT, once) and PV via WMMA
//   3) proj_gemm<f16,1>     : out = PV @ Wo + bo     (f32 flat to d_out)
// ---------------------------------------------------------------------------

typedef __attribute__((ext_vector_type(16))) _Float16 v16h;
typedef __attribute__((ext_vector_type(8)))  float    v8f;
typedef __attribute__((ext_vector_type(4)))  unsigned int u32x4;
typedef __attribute__((ext_vector_type(4)))  int      i32x4;
typedef __attribute__((ext_vector_type(8)))  int      i32x8;

union FragH { v16h v; _Float16 h[16]; unsigned u[8]; };

__device__ __forceinline__ v8f wmma_f16(v16h a, v16h b, v8f c) {
  // D = A(16x32 f16) * B(32x16 f16) + C(16x16 f32)
  return __builtin_amdgcn_wmma_f32_16x16x32_f16(
      /*neg_a=*/false, a, /*neg_b=*/false, b,
      /*c_mod=*/(short)0, c, /*reuse_a=*/false, /*reuse_b=*/false);
}

// A fragment 16x32 (MxK), f16 row-major storage, leading dim ld (halves).
// Lanes 0-15: row M=lane, K pairs {0..7,16..23}; lanes 16-31: K {8..15,24..31}.
__device__ __forceinline__ v16h load_a_frag(const _Float16* base, int ld, int lane) {
  const int m = lane & 15, hh = lane >> 4;
  FragH f;
#pragma unroll
  for (int p = 0; p < 8; ++p) {
    const int k = 2 * (p & 3) + ((p & 4) << 2) + 8 * hh;  // even -> dword load
    f.u[p] = *(const unsigned*)(base + m * ld + k);
  }
  return f.v;
}

// Same A layout, source f32 with on-the-fly cvt (probs in LDS).
__device__ __forceinline__ v16h load_a_frag_f32(const float* base, int ld, int lane) {
  const int m = lane & 15, hh = lane >> 4;
  FragH f;
#pragma unroll
  for (int p = 0; p < 8; ++p) {
    const int k = 2 * (p & 3) + ((p & 4) << 2) + 8 * hh;
    f.h[2 * p]     = (_Float16)base[m * ld + k];
    f.h[2 * p + 1] = (_Float16)base[m * ld + k + 1];
  }
  return f.v;
}

// B fragment 32x16 (KxN), storage holds B^T: element (K,N) at base[n*ld + K].
__device__ __forceinline__ v16h load_bT_frag(const _Float16* base, int ld, int lane) {
  const int n = lane & 15, hh = lane >> 4;
  FragH f;
#pragma unroll
  for (int p = 0; p < 8; ++p) {
    const int k = 2 * (p & 3) + ((p & 4) << 2) + 8 * hh;
    f.u[p] = *(const unsigned*)(base + n * ld + k);
  }
  return f.v;
}

// B fragment 32x16 (KxN), row-major KxN storage: element (K,N) at base[K*ld + n].
__device__ __forceinline__ v16h load_b_frag(const _Float16* base, int ld, int lane) {
  const int n = lane & 15, hh = lane >> 4;
  FragH f;
#pragma unroll
  for (int p = 0; p < 8; ++p) {
    const int k0 = 2 * (p & 3) + ((p & 4) << 2) + 8 * hh;
    f.h[2 * p]     = base[k0 * ld + n];
    f.h[2 * p + 1] = base[(k0 + 1) * ld + n];
  }
  return f.v;
}

// ---------------------------------------------------------------------------
// TDM: DMA one 64x64 f16 tile (row pitch 64 halves) from global memory into
// LDS at byte offset ldsByteOff, hardware-padding +16B after each 128B row so
// LDS rows land on a 144B (KVLD=72 halves) stride. D# per ISA 8.3/8.4:
//   g0: count=1 | lds_addr | global_addr[31:0] | ga[56:32] + type=2
//   g1: data_size=2B, pad_enable, pad_interval=32DW, pad_amount=4DW,
//       tensor_dim0=64, tensor_dim1=2048, tile 64x64, dim0_stride=64
// Issued by one wave; completion via TENSORcnt.
// ---------------------------------------------------------------------------
__device__ __forceinline__ void tdm_load_64x64_f16(unsigned ldsByteOff,
                                                   const _Float16* gsrc) {
  const unsigned long long ga = (unsigned long long)(size_t)gsrc;
  u32x4 g0;
  g0[0] = 1u;                                   // count=1 (user mode)
  g0[1] = ldsByteOff;                           // lds_addr (bytes)
  g0[2] = (unsigned)ga;                         // global_addr[31:0]
  g0[3] = (unsigned)((ga >> 32) & 0x1ffffffull) // global_addr[56:32]
          | 0x80000000u;                        // type=2 ("image")
  i32x8 g1;
  g1[0] = 0x07110000;   // data_size=1(2B), pad_en, pad_interval=4, pad_amount=3
  g1[1] = 0x00400000;   // tensor_dim0 = 64  (bits 79:48, low half)
  g1[2] = 0x08000000;   // tensor_dim1 = 2048 (bits 111:80, low half)
  g1[3] = 0x00400000;   // tile_dim0 = 64   (bits 127:112)
  g1[4] = 0x00000040;   // tile_dim1 = 64   (bits 143:128)
  g1[5] = 0x00000040;   // tensor_dim0_stride = 64 (bits 207:160, low 32)
  g1[6] = 0;
  g1[7] = 0;
  const i32x4 z4 = {0, 0, 0, 0};
#if defined(__clang_major__) && (__clang_major__ >= 23)
  const i32x8 z8 = {0, 0, 0, 0, 0, 0, 0, 0};
  __builtin_amdgcn_tensor_load_to_lds(g0, g1, z4, z4, z8, 0);
#else
  __builtin_amdgcn_tensor_load_to_lds(g0, g1, z4, z4, 0);
#endif
}

// ---------------------------------------------------------------------------
// Projection GEMM: C[M,N] = A[M,K] @ W[K,N] + bias[N]
// Block 128x128, BK=32, 256 threads (8 wave32), wave tile 32x64 (8 WMMAs/chunk).
// A/W are f32 (or f16) and must be converted, so staging goes through VGPRs
// (TDM cannot convert) -> classic LDS tiles here.
// OUT_MODE 0: f16, head layout [B=2,H=16,L=2048,D=64].  OUT_MODE 1: f32 flat.
// ---------------------------------------------------------------------------
template <typename AT, int OUT_MODE>
__global__ __launch_bounds__(256) void proj_gemm(
    const AT* __restrict__ A, const float* __restrict__ W,
    const float* __restrict__ bias, void* __restrict__ outp,
    int M, int N, int K) {
  constexpr int BM = 128, BN = 128, BK = 32;
  constexpr int ALD = BK + 8, BLD = BN + 8;
  __shared__ _Float16 aT[BM * ALD];
  __shared__ _Float16 bT[BK * BLD];

  const int tid = threadIdx.x, lane = tid & 31, wid = tid >> 5;
  const int m0 = blockIdx.y * BM, n0 = blockIdx.x * BN;
  const int waveM = (wid >> 1) * 32, waveN = (wid & 1) * 64;

  v8f zero = {};
  v8f acc[2][4];
#pragma unroll
  for (int i = 0; i < 2; ++i)
#pragma unroll
    for (int j = 0; j < 4; ++j) acc[i][j] = zero;

  for (int k0 = 0; k0 < K; k0 += BK) {
    __syncthreads();
#pragma unroll
    for (int i = 0; i < 16; ++i) {  // 128x32 A tile -> f16
      const int e = tid + 256 * i, r = e >> 5, c = e & 31;
      aT[r * ALD + c] = (_Float16)A[(size_t)(m0 + r) * K + k0 + c];
    }
#pragma unroll
    for (int i = 0; i < 16; ++i) {  // 32x128 W tile -> f16
      const int e = tid + 256 * i, r = e >> 7, c = e & 127;
      bT[r * BLD + c] = (_Float16)W[(size_t)(k0 + r) * N + n0 + c];
    }
    __syncthreads();

    v16h af[2], bf[4];
#pragma unroll
    for (int im = 0; im < 2; ++im)
      af[im] = load_a_frag(aT + (waveM + 16 * im) * ALD, ALD, lane);
#pragma unroll
    for (int jn = 0; jn < 4; ++jn)
      bf[jn] = load_b_frag(bT + waveN + 16 * jn, BLD, lane);
#pragma unroll
    for (int im = 0; im < 2; ++im)
#pragma unroll
      for (int jn = 0; jn < 4; ++jn)
        acc[im][jn] = wmma_f16(af[im], bf[jn], acc[im][jn]);
  }

  const int n = lane & 15, hh = lane >> 4;
#pragma unroll
  for (int im = 0; im < 2; ++im)
#pragma unroll
    for (int jn = 0; jn < 4; ++jn)
#pragma unroll
      for (int r = 0; r < 8; ++r) {
        const int m = m0 + waveM + 16 * im + r + 8 * hh;
        const int nn = n0 + waveN + 16 * jn + n;
        const float val = acc[im][jn][r] + bias[nn];
        if (OUT_MODE == 0) {
          _Float16* o = (_Float16*)outp;
          const int bb = m >> 11, l = m & 2047;  // L = 2048
          const int hd = nn >> 6, d = nn & 63;   // D = 64
          o[(((size_t)(bb * 16 + hd) * 2048) + l) * 64 + d] = (_Float16)val;
        } else {
          float* o = (float*)outp;
          o[(size_t)m * N + nn] = val;
        }
      }
}

// ---------------------------------------------------------------------------
// Fused attention: one block per (b, h, 32 q-rows). 256 threads = 8 wave32,
// waves arranged 2x4 over the 32-row x 64-col working tile.
// Dynamic LDS: 32 x 2052 f32 scores (~257KB) + 2 x 64x72 f16 K/V buffers (18KB).
// K/V tiles arrive via double-buffered TDM DMA (next tile overlaps compute).
// ---------------------------------------------------------------------------
constexpr int SLD = 2052;                           // score row stride (f32)
constexpr int KVLD = 72;                            // K/V row stride (f16, 144B)
constexpr unsigned KV_OFF = 32u * SLD * 4u;         // 262656 B
constexpr unsigned KV_TILE_BYTES = 64u * KVLD * 2u; // 9216 B per buffer

__global__ __launch_bounds__(256) void attn_fused(
    const _Float16* __restrict__ qh, const _Float16* __restrict__ kh,
    const _Float16* __restrict__ vh, const float* __restrict__ bias,
    const unsigned char* __restrict__ mask,
    float* __restrict__ attnOut, _Float16* __restrict__ pvOut) {
  extern __shared__ char smem[];
  float* S = (float*)smem;
  _Float16* kvb0 = (_Float16*)(smem + KV_OFF);
  _Float16* kvb1 = (_Float16*)(smem + KV_OFF + KV_TILE_BYTES);

  const int tid = threadIdx.x, lane = tid & 31, wid = tid >> 5;
  const int b = blockIdx.z, h = blockIdx.y, q0 = blockIdx.x * 32;
  const int waveR = wid >> 2, waveC = wid & 3;
  const int n = lane & 15, hh = lane >> 4;

  const size_t headOff = (size_t)(b * 16 + h) * 2048 * 64;
  const _Float16* kbase = kh + headOff;
  const _Float16* vbase = vh + headOff;

  // Q fragments stay in registers for the whole block (16 rows x D=64 / wave).
  v16h qa[2];
  {
    const _Float16* qbase = qh + headOff + (size_t)(q0 + waveR * 16) * 64;
    qa[0] = load_a_frag(qbase, 64, lane);
    qa[1] = load_a_frag(qbase + 32, 64, lane);
  }

  const float invT = 1.0f / 32.0f;  // 1/sqrt(H*D)

  // ---- Phase 1: S = qh kh^T / T + bias, masked, into LDS -------------------
  if (wid == 0) tdm_load_64x64_f16(KV_OFF, kbase);  // DMA chunk 0 -> buf0
  for (int kc = 0; kc < 2048; kc += 64) {
    const int buf = (kc >> 6) & 1;
    if (wid == 0) __builtin_amdgcn_s_wait_tensorcnt(0);  // tile resident
    __syncthreads();  // publish tile; also fences prev iter's reads of buf^1
    if (wid == 0 && kc + 64 < 2048)  // DMA next tile while we compute this one
      tdm_load_64x64_f16(KV_OFF + (buf ^ 1) * KV_TILE_BYTES,
                         kbase + (size_t)(kc + 64) * 64);
    const _Float16* kv = buf ? kvb1 : kvb0;

    v8f acc = {};
    acc = wmma_f16(qa[0], load_bT_frag(kv + (16 * waveC) * KVLD, KVLD, lane), acc);
    acc = wmma_f16(qa[1], load_bT_frag(kv + (16 * waveC) * KVLD + 32, KVLD, lane), acc);

    const int kg = kc + 16 * waveC + n;  // global k column (== column in S)
#pragma unroll
    for (int r = 0; r < 8; ++r) {
      const int lr = 16 * waveR + 8 * hh + r;  // local q row 0..31
      const int qq = q0 + lr;
      // bias (512MB, single-use) and mask (8MB, reread per head) are streams: non-temporal so they don't
      // evict the K/V working set from the 192MB L2.
      const float bsc = __builtin_nontemporal_load(
          bias + (((size_t)(b * 16 + h) * 2048 + qq) * 2048) + kg);
      const unsigned char mk = __builtin_nontemporal_load(
          mask + ((size_t)b * 2048 + qq) * 2048 + kg);
      float s = acc[r] * invT + bsc;
      if (mk) s = -__builtin_inff();
      S[lr * SLD + kg] = s;
    }
  }

  // ---- Phase 2: exact softmax per row; write attn output (NT) once ---------
  __syncthreads();
  // Overlap V chunk-0 DMA with the whole softmax phase (buf0 is idle now).
  if (wid == 0) tdm_load_64x64_f16(KV_OFF, vbase);
#pragma unroll
  for (int ri = 0; ri < 4; ++ri) {  // 8 waves x 4 rows = 32 rows
    const int lr = wid * 4 + ri;
    float* row = S + lr * SLD;
    float mx = -__builtin_inff();
    for (int c = lane; c < 2048; c += 32) mx = fmaxf(mx, row[c]);
#pragma unroll
    for (int off = 16; off > 0; off >>= 1) mx = fmaxf(mx, __shfl_xor(mx, off, 32));
    float sum = 0.f;
    for (int c = lane; c < 2048; c += 32) {
      const float e = __expf(row[c] - mx);
      row[c] = e;
      sum += e;
    }
#pragma unroll
    for (int off = 16; off > 0; off >>= 1) sum += __shfl_xor(sum, off, 32);
    const float inv = 1.0f / sum;
    float* ao = attnOut + (((size_t)(b * 16 + h)) * 2048 + (q0 + lr)) * 2048;
    for (int c = lane; c < 2048; c += 32) {
      const float p = row[c] * inv;
      row[c] = p;
      __builtin_nontemporal_store(p, ao + c);  // written once, never re-read
    }
  }

  // ---- Phase 3: PV = attn @ vh (f16 operands, f32 accum) -------------------
  v8f oacc = {};
  for (int kc = 0; kc < 2048; kc += 64) {
    const int buf = (kc >> 6) & 1;
    if (wid == 0) __builtin_amdgcn_s_wait_tensorcnt(0);
    __syncthreads();  // iter 0: also publishes softmax'd S to all waves
    if (wid == 0 && kc + 64 < 2048)
      tdm_load_64x64_f16(KV_OFF + (buf ^ 1) * KV_TILE_BYTES,
                         vbase + (size_t)(kc + 64) * 64);
    const _Float16* kv = buf ? kvb1 : kvb0;
#pragma unroll
    for (int kk = 0; kk < 2; ++kk) {
      const v16h pa =
          load_a_frag_f32(S + (16 * waveR) * SLD + kc + 32 * kk, SLD, lane);
      const v16h vb = load_b_frag(kv + (32 * kk) * KVLD + 16 * waveC, KVLD, lane);
      oacc = wmma_f16(pa, vb, oacc);
    }
  }

  // Store PV tile as f16 in [B, L, H*D] layout (re-read by O-proj: keep RT).
#pragma unroll
  for (int r = 0; r < 8; ++r) {
    const int lr = 16 * waveR + 8 * hh + r;
    const int qq = q0 + lr;
    const int d = 16 * waveC + n;
    pvOut[((size_t)(b * 2048 + qq)) * 1024 + h * 64 + d] = (_Float16)oacc[r];
  }
}

// ---------------------------------------------------------------------------
extern "C" void kernel_launch(void* const* d_in, const int* in_sizes, int n_in,
                              void* d_out, int out_size, void* d_ws, size_t ws_size,
                              hipStream_t stream) {
  (void)in_sizes; (void)n_in; (void)out_size; (void)ws_size;

  const float* q = (const float*)d_in[0];
  const float* k = (const float*)d_in[1];
  const float* v = (const float*)d_in[2];
  const unsigned char* mask = (const unsigned char*)d_in[3];  // jnp bool = 1 byte
  const float* bias = (const float*)d_in[4];
  const float* Wq = (const float*)d_in[5];
  const float* bq = (const float*)d_in[6];
  const float* Wk = (const float*)d_in[7];
  const float* bk = (const float*)d_in[8];
  const float* Wv = (const float*)d_in[9];
  const float* bv = (const float*)d_in[10];
  const float* Wo = (const float*)d_in[11];
  const float* bo = (const float*)d_in[12];

  const size_t HEAD_ELEMS = (size_t)2 * 16 * 2048 * 64;  // 4,194,304 f16 each
  _Float16* qh = (_Float16*)d_ws;
  _Float16* kh = qh + HEAD_ELEMS;
  _Float16* vh = kh + HEAD_ELEMS;
  _Float16* pv = vh + HEAD_ELEMS;  // [B, L, H*D] f16

  float* out  = (float*)d_out;                  // [2,2048,1024] f32
  float* attn = out + (size_t)2 * 2048 * 1024;  // [2,16,2048,2048] f32

  const dim3 gGemm(1024 / 128, 4096 / 128);  // (N tiles, M tiles) = (8, 32)
  proj_gemm<float, 0><<<gGemm, 256, 0, stream>>>(q, Wq, bq, (void*)qh, 4096, 1024, 1024);
  proj_gemm<float, 0><<<gGemm, 256, 0, stream>>>(k, Wk, bk, (void*)kh, 4096, 1024, 1024);
  proj_gemm<float, 0><<<gGemm, 256, 0, stream>>>(v, Wv, bv, (void*)vh, 4096, 1024, 1024);

  const dim3 gAttn(2048 / 32, 16, 2);  // (q tiles, H, B) = 2048 workgroups
  const size_t smemBytes = (size_t)KV_OFF + 2 * KV_TILE_BYTES;  // ~275KB of 320KB
  attn_fused<<<gAttn, 256, smemBytes, stream>>>(qh, kh, vh, bias, mask, attn, pv);

  proj_gemm<_Float16, 1><<<gGemm, 256, 0, stream>>>(pv, Wo, bo, d_out, 4096, 1024, 1024);
}
